// OverfitControlPlane_7232724927261
// MI455X (gfx1250) — matrix operand order repaired and checked by reading, not compiled, not run on hardware
//
#include <hip/hip_runtime.h>

typedef __attribute__((ext_vector_type(16))) _Float16 v16h;
typedef __attribute__((ext_vector_type(8)))  _Float16 v8h;
typedef __attribute__((ext_vector_type(8)))  float    v8f;

union AFrag { v16h v; v8h h[2]; };

__device__ __forceinline__ v8f wmma16(v16h a, v16h b, v8f c) {
  return __builtin_amdgcn_wmma_f32_16x16x32_f16(false, a, false, b, (short)0, c, false, false);
}

constexpr int BATCH  = 8;
constexpr int CPD    = 32;
constexpr int FRAMES = 8192;
constexpr int IDIM   = 512;
constexpr int SDIM   = 256;
constexpr int HOP    = 256;
constexpr int CLEN   = 128;   // frames per chunk
constexpr int NCH    = 64;    // chunks

// ---------------- small prep kernels ----------------

// PB[c][s] = sum_i proj[c][i] * IM[i][s]   (32x256)
__global__ void k_pb(const float* __restrict__ proj, const float* __restrict__ IM,
                     float* __restrict__ PB) {
  int c = blockIdx.x, s = threadIdx.x;
  float a = 0.f;
  for (int i = 0; i < IDIM; ++i) a += proj[c * IDIM + i] * IM[i * SDIM + s];
  PB[c * SDIM + s] = a;
}

// PD[c][o] = sum_i proj[c][i] * DM[i][o]   (32x512)
__global__ void k_pd(const float* __restrict__ proj, const float* __restrict__ DM,
                     float* __restrict__ PD) {
  int c = blockIdx.x, o = threadIdx.x;
  float a = 0.f;
  for (int i = 0; i < IDIM; ++i) a += proj[c * IDIM + i] * DM[i * IDIM + o];
  PD[c * IDIM + o] = a;
}

// Pre-shuffle a row-major f32 (256 x Ndim) matrix into B-fragment-major f16:
// frag[((ntile*8 + k)*32 + lane)*16 + h] = W[K][N],
//   K = 32k + 8*(lane>>4) + (h&7) + 16*(h>>3),  N = ntile*16 + (lane&15)
__global__ void k_fragB(const float* __restrict__ W, _Float16* __restrict__ frag,
                        int Ndim, int total) {
  int idx = blockIdx.x * blockDim.x + threadIdx.x;
  if (idx >= total) return;
  int h = idx & 15;
  int lane = (idx >> 4) & 31;
  int k = (idx >> 9) & 7;
  int ntile = idx >> 12;
  int K = (k << 5) + ((lane >> 4) << 3) + (h & 7) + ((h >> 3) << 4);
  int N = (ntile << 4) + (lane & 15);
  frag[idx] = (_Float16)W[K * Ndim + N];
}

// f32 -> f16 row-major copy
__global__ void k_cvt16(const float* __restrict__ src, _Float16* __restrict__ dst, int total) {
  int i = blockIdx.x * blockDim.x + threadIdx.x;
  if (i < total) dst[i] = (_Float16)src[i];
}

// Mout = Min @ Min (256x256, f16 in, f32 acc, f16 out). grid=16 row slabs, 512 thr.
__global__ __launch_bounds__(512) void k_sq(const _Float16* __restrict__ Min,
                                            _Float16* __restrict__ Mout) {
  int w = threadIdx.x >> 5, lane = threadIdx.x & 31;
  int col   = (w << 4) + (lane & 15);
  int row   = (blockIdx.x << 4) + (lane & 15);
  int koff  = (lane >> 4) << 3;
  int rbase = (blockIdx.x << 4) + ((lane >> 4) << 3);
  v16h bf[8];
#pragma unroll
  for (int k = 0; k < 8; ++k) {
#pragma unroll
    for (int h = 0; h < 16; ++h)
      bf[k][h] = Min[(((k << 5) + ((lane >> 4) << 3) + (h & 7) + ((h >> 3) << 4))) * SDIM + col];
  }
  v8f acc = {};
#pragma unroll
  for (int k = 0; k < 8; ++k) {
    AFrag a;
    a.h[0] = *(const v8h*)&Min[row * SDIM + (k << 5) + koff];
    a.h[1] = *(const v8h*)&Min[row * SDIM + (k << 5) + koff + 16];
    acc = wmma16(a.v, bf[k], acc);
  }
#pragma unroll
  for (int v = 0; v < 8; ++v) Mout[(rbase + v) * SDIM + col] = (_Float16)acc[v];
}

// Front-end, batch-innermost layouts:
// ib[f][s][b] = sum_c sig[b][c][f]*PB[c][s];  id[f][o][b] = sum_c sig[b][c][f]*PD[c][o]
__global__ void k_projsig(const float* __restrict__ sig, const float* __restrict__ PB,
                          const float* __restrict__ PD, float* __restrict__ ib,
                          float* __restrict__ idb) {
  int f = blockIdx.x, t = threadIdx.x;
  __shared__ float sg[BATCH][CPD];
  if (t < BATCH * CPD) {
    int b = t >> 5, c = t & 31;
    sg[b][c] = sig[((size_t)b * CPD + c) * FRAMES + f];
  }
  __syncthreads();
  v8f av = {}, d0v = {}, d1v = {};
#pragma unroll 4
  for (int c = 0; c < CPD; ++c) {
    float pb  = PB[c * SDIM + t];
    float pd0 = PD[c * IDIM + t];
    float pd1 = PD[c * IDIM + t + 256];
#pragma unroll
    for (int b = 0; b < BATCH; ++b) {
      float s = sg[b][c];
      av[b]  += s * pb;
      d0v[b] += s * pd0;
      d1v[b] += s * pd1;
    }
  }
  *(v8f*)(ib  + ((size_t)f * SDIM + t) * BATCH)        = av;
  *(v8f*)(idb + ((size_t)f * IDIM + t) * BATCH)        = d0v;
  *(v8f*)(idb + ((size_t)f * IDIM + t + 256) * BATCH)  = d1v;
}

// ---------------- chunked scan ----------------

// Pass 1: local states per chunk (zero initial state).  grid=NCH, 512 thr (16 waves).
__global__ __launch_bounds__(512) void k_scan_local(const float* __restrict__ ib,
                                                    const _Float16* __restrict__ fragWs,
                                                    float* __restrict__ ls) {
  __shared__ __attribute__((aligned(16))) _Float16 st[16 * SDIM];
  int chunk = blockIdx.x;
  int tid = threadIdx.x, w = tid >> 5, lane = tid & 31;
  int col = (w << 4) + (lane & 15);
  int koff = (lane >> 4) << 3;
  int rbase = (lane >> 4) << 3;
  int rowA = (lane & 15) << 8;     // row*256
  v16h bf[8];
#pragma unroll
  for (int k = 0; k < 8; ++k) bf[k] = *(const v16h*)&fragWs[((w * 8 + k) * 32 + lane) << 4];
  for (int i = tid; i < 16 * SDIM; i += 512) st[i] = (_Float16)0.f;
  __syncthreads();
  for (int fi = 0; fi < CLEN; ++fi) {
    int f = chunk * CLEN + fi;
    // batch-innermost: 8 consecutive floats per lane, rows 0..7 only
    v8f ibv = {};
    if (lane < 16) {
      ibv = *(const v8f*)(ib + ((size_t)f * SDIM + col) * BATCH);
      __builtin_prefetch(ib + ((size_t)(f + 1) * SDIM + col) * BATCH, 0, 0);
    }
    v8f acc = {};
#pragma unroll
    for (int k = 0; k < 8; ++k) {
      AFrag a;
      a.h[0] = *(const v8h*)&st[rowA + (k << 5) + koff];
      a.h[1] = *(const v8h*)&st[rowA + (k << 5) + koff + 16];
      acc = wmma16(a.v, bf[k], acc);
    }
    float vals[8];
#pragma unroll
    for (int v = 0; v < 8; ++v) vals[v] = acc[v] + ((rbase + v < 8) ? ibv[v] : 0.f);
    __syncthreads();
#pragma unroll
    for (int v = 0; v < 8; ++v) st[(rbase + v) * SDIM + col] = (_Float16)vals[v];
    if (lane < 16) {    // rbase==0, rows 0..7 are real batches
      v8f y;
#pragma unroll
      for (int v = 0; v < 8; ++v) y[v] = vals[v];
      *(v8f*)(ls + ((size_t)f * SDIM + col) * BATCH) = y;
    }
    __syncthreads();
  }
}

// Prefix over chunk carries: c_{j+1} = c_j @ A^128 + ls_end_j.  1 block, 512 thr.
__global__ __launch_bounds__(512) void k_prefix(const float* __restrict__ ls,
                                                const _Float16* __restrict__ A128,
                                                float* __restrict__ carries) {
  __shared__ __attribute__((aligned(16))) _Float16 st[16 * SDIM];
  int tid = threadIdx.x, w = tid >> 5, lane = tid & 31;
  int col = (w << 4) + (lane & 15);
  int koff = (lane >> 4) << 3;
  int rbase = (lane >> 4) << 3;
  int rowA = (lane & 15) << 8;
  v16h bf[8];
#pragma unroll
  for (int k = 0; k < 8; ++k) {
#pragma unroll
    for (int h = 0; h < 16; ++h)
      bf[k][h] = A128[(((k << 5) + ((lane >> 4) << 3) + (h & 7) + ((h >> 3) << 4))) * SDIM + col];
  }
  float cv[8];
#pragma unroll
  for (int v = 0; v < 8; ++v) cv[v] = 0.f;
  for (int j = 0; j < NCH; ++j) {
    if (lane < 16) {
      v8f y;
#pragma unroll
      for (int v = 0; v < 8; ++v) y[v] = cv[v];
      *(v8f*)(carries + ((size_t)j * SDIM + col) * BATCH) = y;
    }
#pragma unroll
    for (int v = 0; v < 8; ++v) st[(rbase + v) * SDIM + col] = (_Float16)cv[v];
    __syncthreads();
    v8f acc = {};
#pragma unroll
    for (int k = 0; k < 8; ++k) {
      AFrag a;
      a.h[0] = *(const v8h*)&st[rowA + (k << 5) + koff];
      a.h[1] = *(const v8h*)&st[rowA + (k << 5) + koff + 16];
      acc = wmma16(a.v, bf[k], acc);
    }
    v8f lse = {};
    if (lane < 16)
      lse = *(const v8f*)(ls + ((size_t)(j * CLEN + (CLEN - 1)) * SDIM + col) * BATCH);
    __syncthreads();
#pragma unroll
    for (int v = 0; v < 8; ++v) cv[v] = acc[v] + ((rbase + v < 8) ? lse[v] : 0.f);
  }
}

// Pass 2: state_f = ls_f + carry_f, carry_f = carry_{f-1} @ A.  Writes f16 states (16 rows).
__global__ __launch_bounds__(512) void k_fixup(const float* __restrict__ ls,
                                               const float* __restrict__ carries,
                                               const _Float16* __restrict__ fragWs,
                                               _Float16* __restrict__ st16) {
  __shared__ __attribute__((aligned(16))) _Float16 st[16 * SDIM];
  int chunk = blockIdx.x;
  int tid = threadIdx.x, w = tid >> 5, lane = tid & 31;
  int col = (w << 4) + (lane & 15);
  int koff = (lane >> 4) << 3;
  int rbase = (lane >> 4) << 3;
  int rowA = (lane & 15) << 8;
  v16h bf[8];
#pragma unroll
  for (int k = 0; k < 8; ++k) bf[k] = *(const v16h*)&fragWs[((w * 8 + k) * 32 + lane) << 4];
  const float* carr = carries + (size_t)chunk * SDIM * BATCH;   // [s][b]
  for (int i = tid; i < 16 * SDIM; i += 512) {
    int r = i >> 8, c = i & 255;
    st[i] = (r < 8) ? (_Float16)carr[c * BATCH + r] : (_Float16)0.f;
  }
  __syncthreads();
  for (int fi = 0; fi < CLEN; ++fi) {
    int f = chunk * CLEN + fi;
    v8f lsv = {};
    if (lane < 16) {
      lsv = *(const v8f*)(ls + ((size_t)f * SDIM + col) * BATCH);
      __builtin_prefetch(ls + ((size_t)(f + 1) * SDIM + col) * BATCH, 0, 0);
    }
    v8f acc = {};
#pragma unroll
    for (int k = 0; k < 8; ++k) {
      AFrag a;
      a.h[0] = *(const v8h*)&st[rowA + (k << 5) + koff];
      a.h[1] = *(const v8h*)&st[rowA + (k << 5) + koff + 16];
      acc = wmma16(a.v, bf[k], acc);
    }
    _Float16* sf = st16 + (size_t)f * 16 * SDIM;
    __syncthreads();
#pragma unroll
    for (int v = 0; v < 8; ++v) {
      int r = rbase + v;
      float stv = acc[v] + ((r < 8) ? lsv[v] : 0.f);
      sf[r * SDIM + col] = (_Float16)stv;           // total state (f16) for output GEMM
      st[r * SDIM + col] = (_Float16)acc[v];        // carry for next step
    }
    __syncthreads();
  }
}

// out_f = state_f @ Wout + id_f  (in-place accumulate into id buffer, [f][o][b] layout).
__global__ __launch_bounds__(512) void k_out(const _Float16* __restrict__ st16,
                                             const _Float16* __restrict__ fragWo,
                                             float* __restrict__ io) {
  int f = blockIdx.x;
  int tid = threadIdx.x, w = tid >> 5, lane = tid & 31;
  int n0 = 2 * w;
  int col0 = (n0 << 4) + (lane & 15);
  int koff = (lane >> 4) << 3;
  int rowA = (lane & 15) << 8;
  v16h b0[8], b1[8];
#pragma unroll
  for (int k = 0; k < 8; ++k) {
    b0[k] = *(const v16h*)&fragWo[((n0 * 8 + k) * 32 + lane) << 4];
    b1[k] = *(const v16h*)&fragWo[(((n0 + 1) * 8 + k) * 32 + lane) << 4];
  }
  const _Float16* A = st16 + (size_t)f * 16 * SDIM;
  v8f a0 = {}, a1 = {};
#pragma unroll
  for (int k = 0; k < 8; ++k) {
    AFrag a;
    a.h[0] = *(const v8h*)&A[rowA + (k << 5) + koff];
    a.h[1] = *(const v8h*)&A[rowA + (k << 5) + koff + 16];
    a0 = wmma16(a.v, b0[k], a0);
    a1 = wmma16(a.v, b1[k], a1);
  }
  if (lane < 16) {   // rows 0..7 = real batches, 8 consecutive floats per column
    float* of = io + (size_t)f * IDIM * BATCH;
    v8f* p0 = (v8f*)(of + (size_t)col0 * BATCH);
    v8f* p1 = (v8f*)(of + (size_t)(col0 + 16) * BATCH);
    v8f x0 = *p0, x1 = *p1;
#pragma unroll
    for (int v = 0; v < 8; ++v) { x0[v] += a0[v]; x1[v] += a1[v]; }
    *p0 = x0;
    *p1 = x1;
  }
}

// overlap-add from [f][o][b]: out[b][f*256+t] = ob[f][t][b] + ob[f-1][256+t][b]
__global__ void k_ola(const float* __restrict__ ob, float* __restrict__ out) {
  int f = blockIdx.x, t = threadIdx.x;
  v8f cur = *(const v8f*)(ob + ((size_t)f * IDIM + t) * BATCH);
  v8f prev = {};
  if (f > 0) prev = *(const v8f*)(ob + ((size_t)(f - 1) * IDIM + 256 + t) * BATCH);
#pragma unroll
  for (int b = 0; b < BATCH; ++b)
    out[(size_t)b * ((size_t)FRAMES * HOP) + (size_t)f * HOP + t] = cur[b] + prev[b];
}

// ---------------- host ----------------

extern "C" void kernel_launch(void* const* d_in, const int* in_sizes, int n_in,
                              void* d_out, int out_size, void* d_ws, size_t ws_size,
                              hipStream_t stream) {
  const float* sig  = (const float*)d_in[0];
  const float* proj = (const float*)d_in[1];
  const float* SM   = (const float*)d_in[2];
  const float* IM   = (const float*)d_in[3];
  const float* OM   = (const float*)d_in[4];
  const float* DM   = (const float*)d_in[5];
  float* out = (float*)d_out;

  char* p = (char*)d_ws;
  auto carve = [&](size_t bytes) -> char* {
    char* r = p;
    p += (bytes + 255) & ~(size_t)255;
    return r;
  };
  float*    PB      = (float*)   carve((size_t)CPD * SDIM * 4);
  float*    PD      = (float*)   carve((size_t)CPD * IDIM * 4);
  _Float16* fragWs  = (_Float16*)carve((size_t)16 * 8 * 32 * 16 * 2);
  _Float16* fragWo  = (_Float16*)carve((size_t)32 * 8 * 32 * 16 * 2);
  _Float16* A16a    = (_Float16*)carve((size_t)SDIM * SDIM * 2);
  _Float16* A16b    = (_Float16*)carve((size_t)SDIM * SDIM * 2);
  float*    carries = (float*)   carve((size_t)NCH * SDIM * BATCH * 4);
  float*    ib      = (float*)   carve((size_t)FRAMES * SDIM * BATCH * 4);
  float*    ls      = (float*)   carve((size_t)FRAMES * SDIM * BATCH * 4);
  _Float16* st16    = (_Float16*)carve((size_t)FRAMES * 16 * SDIM * 2);
  float*    idout   = (float*)   carve((size_t)FRAMES * IDIM * BATCH * 4);

  // folded projection matrices
  k_pb<<<CPD, SDIM, 0, stream>>>(proj, IM, PB);
  k_pd<<<CPD, IDIM, 0, stream>>>(proj, DM, PD);
  // fragment-major f16 copies of state & output matrices
  k_fragB<<<(16 * 8 * 32 * 16) / 256, 256, 0, stream>>>(SM, fragWs, SDIM, 16 * 8 * 32 * 16);
  k_fragB<<<(32 * 8 * 32 * 16) / 256, 256, 0, stream>>>(OM, fragWo, IDIM, 32 * 8 * 32 * 16);
  // A^128 via 7 WMMA squarings
  k_cvt16<<<(SDIM * SDIM) / 256, 256, 0, stream>>>(SM, A16a, SDIM * SDIM);
  k_sq<<<16, 512, 0, stream>>>(A16a, A16b);  // A^2
  k_sq<<<16, 512, 0, stream>>>(A16b, A16a);  // A^4
  k_sq<<<16, 512, 0, stream>>>(A16a, A16b);  // A^8
  k_sq<<<16, 512, 0, stream>>>(A16b, A16a);  // A^16
  k_sq<<<16, 512, 0, stream>>>(A16a, A16b);  // A^32
  k_sq<<<16, 512, 0, stream>>>(A16b, A16a);  // A^64
  k_sq<<<16, 512, 0, stream>>>(A16a, A16b);  // A^128 -> A16b
  // front-end: ib / id (id written into the out buffer, accumulated in-place later)
  k_projsig<<<FRAMES, 256, 0, stream>>>(sig, PB, PD, ib, idout);
  // chunked scan
  k_scan_local<<<NCH, 512, 0, stream>>>(ib, fragWs, ls);
  k_prefix<<<1, 512, 0, stream>>>(ls, A16b, carries);
  k_fixup<<<NCH, 512, 0, stream>>>(ls, carries, fragWs, st16);
  // output GEMM + overlap-add
  k_out<<<FRAMES, 512, 0, stream>>>(st16, fragWo, idout);
  k_ola<<<FRAMES, 256, 0, stream>>>(idout, out);

  (void)in_sizes; (void)n_in; (void)out_size; (void)ws_size;
}
